// DualGNNInterface_24833500905822
// MI455X (gfx1250) — matrix operand order repaired and checked by reading, not compiled, not run on hardware
//
#include <hip/hip_runtime.h>

// ---------------------------------------------------------------------------
// Types for CDNA5 WMMA (gfx1250, wave32)
// ---------------------------------------------------------------------------
typedef __bf16 bf16_t;
typedef bf16_t v16bf __attribute__((ext_vector_type(16)));
typedef bf16_t v8bf  __attribute__((ext_vector_type(8)));
typedef float  v8f   __attribute__((ext_vector_type(8)));

#define C_DIM 256
#define NIMG  65536
#define NCLS  4096
#define IMG_B 4
#define IMG_H 128
#define IMG_W 128

__device__ __forceinline__ float wave_sum32(float v) {
#pragma unroll
  for (int m = 16; m >= 1; m >>= 1) v += __shfl_xor(v, m, 32);
  return v;
}

__device__ __forceinline__ void atomic_add_f32(float* p, float v) {
  __hip_atomic_fetch_add(p, v, __ATOMIC_RELAXED, __HIP_MEMORY_SCOPE_AGENT);
}

// ---------------------------------------------------------------------------
// [B,C,H,W] f32 -> node-major [N, C] f32 + bf16   (LDS tiled transpose)
// grid: (W/32, C/32, B*H), block (32,8)
// ---------------------------------------------------------------------------
__global__ __launch_bounds__(256) void k_nchw_to_nodes(
    const float* __restrict__ x, float* __restrict__ xf, bf16_t* __restrict__ xb) {
  __shared__ float tile[32][33];
  const int tx = threadIdx.x, ty = threadIdx.y;
  const int wt = blockIdx.x * 32, ct = blockIdx.y * 32;
  const int bh = blockIdx.z;               // b*H + h
  const int b = bh / IMG_H, h = bh % IMG_H;
#pragma unroll
  for (int i = 0; i < 4; ++i) {
    int c = ct + ty + i * 8;
    tile[ty + i * 8][tx] =
        x[(((size_t)b * C_DIM + c) * IMG_H + h) * IMG_W + wt + tx];
  }
  __syncthreads();
#pragma unroll
  for (int i = 0; i < 4; ++i) {
    int node = ((b * IMG_H + h) * IMG_W) + wt + (ty + i * 8);
    int c = ct + tx;
    float v = tile[tx][ty + i * 8];
    xf[(size_t)node * C_DIM + c] = v;
    xb[(size_t)node * C_DIM + c] = (bf16_t)v;
  }
}

// node-major [N, C] f32 -> [B,C,H,W] f32 (same tiling, inverse direction)
__global__ __launch_bounds__(256) void k_nodes_to_nchw(
    const float* __restrict__ xn, float* __restrict__ out) {
  __shared__ float tile[32][33];
  const int tx = threadIdx.x, ty = threadIdx.y;
  const int wt = blockIdx.x * 32, ct = blockIdx.y * 32;
  const int bh = blockIdx.z;
  const int b = bh / IMG_H, h = bh % IMG_H;
#pragma unroll
  for (int i = 0; i < 4; ++i) {
    int node = ((b * IMG_H + h) * IMG_W) + wt + (ty + i * 8);
    tile[ty + i * 8][tx] = xn[(size_t)node * C_DIM + ct + tx];   // tile[w][c]
  }
  __syncthreads();
#pragma unroll
  for (int i = 0; i < 4; ++i) {
    int c = ct + ty + i * 8;
    out[(((size_t)b * C_DIM + c) * IMG_H + h) * IMG_W + wt + tx] =
        tile[tx][ty + i * 8];
  }
}

// class_x f32 -> xc master f32 + bf16
__global__ void k_cls_init(const float* __restrict__ src, float* __restrict__ xf,
                           bf16_t* __restrict__ xb) {
  size_t i = (size_t)blockIdx.x * blockDim.x + threadIdx.x;
  float v = src[i];
  xf[i] = v;
  xb[i] = (bf16_t)v;
}

// ---------------------------------------------------------------------------
// edm2 weight renorm: one wave handles one row of Wl AND one row of Wr.
// grid: 6*256/8 = 192 blocks of 256 threads (8 waves).
// scale = 1 / ((eps + ||w|| / 16) * 16),  s = sqrt(fan_in) = 16
// ---------------------------------------------------------------------------
__device__ __forceinline__ void renorm_row(const float* __restrict__ src,
                                           bf16_t* __restrict__ dst, int lane) {
  float v[8];
  float s2 = 0.f;
#pragma unroll
  for (int i = 0; i < 8; ++i) {
    v[i] = src[lane * 8 + i];
    s2 += v[i] * v[i];
  }
  s2 = wave_sum32(s2);
  float n = sqrtf(s2);
  float scale = 1.0f / ((1e-4f + n * (1.0f / 16.0f)) * 16.0f);
#pragma unroll
  for (int i = 0; i < 8; ++i) dst[lane * 8 + i] = (bf16_t)(v[i] * scale);
}

__global__ __launch_bounds__(256) void k_renorm(
    const float* __restrict__ Wl, const float* __restrict__ Wr,
    bf16_t* __restrict__ wout) {
  int gw = (blockIdx.x * 256 + threadIdx.x) >> 5;   // 0..1535 (6 mats * 256 rows)
  int lane = threadIdx.x & 31;
  int mat = gw >> 8, row = gw & 255;
  size_t src_off = ((size_t)mat * 256 + row) * 256;
  renorm_row(Wl + src_off, wout + ((size_t)(mat * 2 + 0) * 256 + row) * 256, lane);
  renorm_row(Wr + src_off, wout + ((size_t)(mat * 2 + 1) * 256 + row) * 256, lane);
}

// Wout_img + Wout_cls -> bf16 (no renorm per reference)
__global__ void k_wout_cvt(const float* __restrict__ wi, const float* __restrict__ wc,
                           bf16_t* __restrict__ dst) {
  size_t i = (size_t)blockIdx.x * blockDim.x + threadIdx.x;
  if (i < 65536) dst[i] = (bf16_t)wi[i];
  else           dst[i] = (bf16_t)wc[i - 65536];
}

// ---------------------------------------------------------------------------
// Edge scatter: one wave per edge, 8 floats/lane (256 channels),
// float atomics into acc + per-dst count.
// ---------------------------------------------------------------------------
__global__ __launch_bounds__(256) void k_scatter(
    const float* __restrict__ src, const int* __restrict__ esrc,
    const int* __restrict__ edst, int E, float* __restrict__ acc,
    float* __restrict__ cnt) {
  int e = (int)(((size_t)blockIdx.x * blockDim.x + threadIdx.x) >> 5);
  int lane = threadIdx.x & 31;
  if (e >= E) return;
  int s = esrc[e], d = edst[e];
  const float4* sp = (const float4*)(src + (size_t)s * C_DIM + lane * 8);
  float4 v0 = sp[0], v1 = sp[1];
  float* ap = acc + (size_t)d * C_DIM + lane * 8;
  atomic_add_f32(ap + 0, v0.x); atomic_add_f32(ap + 1, v0.y);
  atomic_add_f32(ap + 2, v0.z); atomic_add_f32(ap + 3, v0.w);
  atomic_add_f32(ap + 4, v1.x); atomic_add_f32(ap + 5, v1.y);
  atomic_add_f32(ap + 6, v1.z); atomic_add_f32(ap + 7, v1.w);
  if (lane == 0) atomic_add_f32(cnt + d, 1.0f);
}

// mean = acc / max(cnt,1) -> bf16, one block per row
__global__ __launch_bounds__(256) void k_mean_bf16(
    const float* __restrict__ acc, const float* __restrict__ cnt,
    bf16_t* __restrict__ mean) {
  int row = blockIdx.x;
  float inv = 1.0f / fmaxf(cnt[row], 1.0f);
  size_t i = (size_t)row * C_DIM + threadIdx.x;
  mean[i] = (bf16_t)(acc[i] * inv);
}

// ---------------------------------------------------------------------------
// WMMA GEMM:  out[n,o] = bias[o] + A1[n,:]·B1[o,:] + (A2 ? A2[n,:]·B2[o,:] : 0)
// A* : [N,256] bf16 row-major,  B* : [256(o),256(k)] bf16 row-major.
//
// Register-blocked 2x2: each wave owns a 32x32 output tile (4 accumulators);
// per k-step, 2 A-frags + 2 B-frags feed 4 v_wmma  (1 frag load per WMMA).
// Block = 8 waves covering the full 256-wide output dim; grid = N/32 blocks.
// Fragment layouts: k = kb + 16*lhi + s in A and B (A: 2x16B runs/lane; B: column-major
// per lane -> one 32B contiguous run of the weight row).
// ---------------------------------------------------------------------------
__device__ __forceinline__ v16bf load_A_frag(const bf16_t* __restrict__ rowp,
                                             int kb, int lhi) {
  const bf16_t* p = rowp + kb + lhi * 16;
  v8bf lo = *(const v8bf*)p;          // K = kb+16*lhi .. +7
  v8bf hi = *(const v8bf*)(p + 8);    // K = +8
  v16bf r;
#pragma unroll
  for (int i = 0; i < 8; ++i) { r[i] = lo[i]; r[i + 8] = hi[i]; }
  return r;
}

__device__ __forceinline__ v16bf load_B_frag(const bf16_t* __restrict__ W,
                                             int o, int kb, int lhi) {
  return *(const v16bf*)(W + (size_t)o * C_DIM + kb + lhi * 16);
}

__global__ __launch_bounds__(256) void k_gemm_sage(
    const bf16_t* __restrict__ A1, const bf16_t* __restrict__ A2,
    const bf16_t* __restrict__ B1, const bf16_t* __restrict__ B2,
    const float* __restrict__ bias, float* __restrict__ out) {
  const int lane = threadIdx.x & 31;
  const int wave = threadIdx.x >> 5;
  const int row_base = blockIdx.x * 32;   // 2 row tiles per wave
  const int obase = wave * 32;            // 2 col tiles per wave; 8 waves = 256
  const int lhi = lane >> 4;
  const int lmod = lane & 15;

  float bv0 = bias[obase + lmod];
  float bv1 = bias[obase + 16 + lmod];
  v8f c00, c01, c10, c11;
#pragma unroll
  for (int i = 0; i < 8; ++i) { c00[i] = bv0; c10[i] = bv0; c01[i] = bv1; c11[i] = bv1; }

  {
    const bf16_t* ar0 = A1 + (size_t)(row_base + lmod) * C_DIM;
    const bf16_t* ar1 = A1 + (size_t)(row_base + 16 + lmod) * C_DIM;
#pragma unroll
    for (int ks = 0; ks < 8; ++ks) {
      int kb = ks * 32;
      v16bf a0 = load_A_frag(ar0, kb, lhi);
      v16bf a1 = load_A_frag(ar1, kb, lhi);
      v16bf b0 = load_B_frag(B1, obase + lmod, kb, lhi);
      v16bf b1 = load_B_frag(B1, obase + 16 + lmod, kb, lhi);
      c00 = __builtin_amdgcn_wmma_f32_16x16x32_bf16(false, a0, false, b0, (short)0, c00, false, false);
      c01 = __builtin_amdgcn_wmma_f32_16x16x32_bf16(false, a0, false, b1, (short)0, c01, false, false);
      c10 = __builtin_amdgcn_wmma_f32_16x16x32_bf16(false, a1, false, b0, (short)0, c10, false, false);
      c11 = __builtin_amdgcn_wmma_f32_16x16x32_bf16(false, a1, false, b1, (short)0, c11, false, false);
    }
  }
  if (A2) {
    const bf16_t* ar0 = A2 + (size_t)(row_base + lmod) * C_DIM;
    const bf16_t* ar1 = A2 + (size_t)(row_base + 16 + lmod) * C_DIM;
#pragma unroll
    for (int ks = 0; ks < 8; ++ks) {
      int kb = ks * 32;
      v16bf a0 = load_A_frag(ar0, kb, lhi);
      v16bf a1 = load_A_frag(ar1, kb, lhi);
      v16bf b0 = load_B_frag(B2, obase + lmod, kb, lhi);
      v16bf b1 = load_B_frag(B2, obase + 16 + lmod, kb, lhi);
      c00 = __builtin_amdgcn_wmma_f32_16x16x32_bf16(false, a0, false, b0, (short)0, c00, false, false);
      c01 = __builtin_amdgcn_wmma_f32_16x16x32_bf16(false, a0, false, b1, (short)0, c01, false, false);
      c10 = __builtin_amdgcn_wmma_f32_16x16x32_bf16(false, a1, false, b0, (short)0, c10, false, false);
      c11 = __builtin_amdgcn_wmma_f32_16x16x32_bf16(false, a1, false, b1, (short)0, c11, false, false);
    }
  }

  // C/D layout: VGPR r -> M = r + 8*lhi ; N = lane&15
  {
    float* op00 = out + (size_t)(row_base + lhi * 8) * C_DIM + obase + lmod;
    float* op01 = op00 + 16;
    float* op10 = out + (size_t)(row_base + 16 + lhi * 8) * C_DIM + obase + lmod;
    float* op11 = op10 + 16;
#pragma unroll
    for (int r = 0; r < 8; ++r) {
      op00[(size_t)r * C_DIM] = c00[r];
      op01[(size_t)r * C_DIM] = c01[r];
      op10[(size_t)r * C_DIM] = c10[r];
      op11[(size_t)r * C_DIM] = c11[r];
    }
  }
}

// ---------------------------------------------------------------------------
// Row-wise: h = A/max(||A||,1e-12) + B/max(||B||,1e-12); x = mp_silu(h).
// One block (256 thr) per row; writes f32 master + bf16 copy.
// ---------------------------------------------------------------------------
__device__ __forceinline__ float mp_silu_f(float h) {
  return h * (1.0f / (1.0f + __expf(-h))) * (1.0f / 0.596f);
}

__global__ __launch_bounds__(256) void k_combine2(
    const float* __restrict__ A, const float* __restrict__ B,
    float* __restrict__ xf, bf16_t* __restrict__ xb) {
  __shared__ float pa[8], pb[8];
  int row = blockIdx.x;
  size_t i = (size_t)row * C_DIM + threadIdx.x;
  float a = A[i], b = B[i];
  float sa = wave_sum32(a * a), sb = wave_sum32(b * b);
  int w = threadIdx.x >> 5, l = threadIdx.x & 31;
  if (l == 0) { pa[w] = sa; pb[w] = sb; }
  __syncthreads();
  if (threadIdx.x == 0) {
    float ta = 0.f, tb = 0.f;
#pragma unroll
    for (int j = 0; j < 8; ++j) { ta += pa[j]; tb += pb[j]; }
    pa[0] = ta; pb[0] = tb;
  }
  __syncthreads();
  float na = fmaxf(sqrtf(pa[0]), 1e-12f);
  float nb = fmaxf(sqrtf(pb[0]), 1e-12f);
  float h = a / na + b / nb;
  float v = mp_silu_f(h);
  xf[i] = v;
  xb[i] = (bf16_t)v;
}

__global__ __launch_bounds__(256) void k_combine1(
    const float* __restrict__ A, float* __restrict__ xf, bf16_t* __restrict__ xb) {
  __shared__ float pa[8];
  int row = blockIdx.x;
  size_t i = (size_t)row * C_DIM + threadIdx.x;
  float a = A[i];
  float sa = wave_sum32(a * a);
  int w = threadIdx.x >> 5, l = threadIdx.x & 31;
  if (l == 0) pa[w] = sa;
  __syncthreads();
  if (threadIdx.x == 0) {
    float ta = 0.f;
#pragma unroll
    for (int j = 0; j < 8; ++j) ta += pa[j];
    pa[0] = ta;
  }
  __syncthreads();
  float na = fmaxf(sqrtf(pa[0]), 1e-12f);
  float v = mp_silu_f(a / na);
  xf[i] = v;
  xb[i] = (bf16_t)v;
}

// ---------------------------------------------------------------------------
// Host orchestration
// ---------------------------------------------------------------------------
extern "C" void kernel_launch(void* const* d_in, const int* in_sizes, int n_in,
                              void* d_out, int out_size, void* d_ws, size_t ws_size,
                              hipStream_t stream) {
  (void)n_in; (void)out_size; (void)ws_size;
  const float* x        = (const float*)d_in[0];
  const float* class_x  = (const float*)d_in[1];
  const int*   ei_ii    = (const int*)d_in[2];
  const int*   ei_ci    = (const int*)d_in[3];
  const int*   ei_ic    = (const int*)d_in[4];
  const float* Wl       = (const float*)d_in[5];
  const float* bl       = (const float*)d_in[6];
  const float* Wr       = (const float*)d_in[7];
  const float* Wout_img = (const float*)d_in[8];
  const float* bout_img = (const float*)d_in[9];
  const float* Wout_cls = (const float*)d_in[10];
  const float* bout_cls = (const float*)d_in[11];

  const int E_II = in_sizes[2] / 2;
  const int E_CI = in_sizes[3] / 2;
  const int E_IC = in_sizes[4] / 2;

  // ---- workspace layout (bytes) ----
  char* ws = (char*)d_ws;
  const size_t SZ_IMG_F32 = (size_t)NIMG * C_DIM * 4;   // 64 MB
  const size_t SZ_CLS_F32 = (size_t)NCLS * C_DIM * 4;   //  4 MB
  const size_t SZ_IMG_BF  = (size_t)NIMG * C_DIM * 2;   // 32 MB
  const size_t SZ_CLS_BF  = (size_t)NCLS * C_DIM * 2;   //  2 MB
  size_t off = 0;
  float*  XI_F32 = (float*)(ws + off);  off += SZ_IMG_F32;
  float*  XC_F32 = (float*)(ws + off);  off += SZ_CLS_F32;
  bf16_t* XI_BF  = (bf16_t*)(ws + off); off += SZ_IMG_BF;
  bf16_t* XC_BF  = (bf16_t*)(ws + off); off += SZ_CLS_BF;
  float*  ACC    = (float*)(ws + off);  off += SZ_IMG_F32;  // also reused as OUTB
  float*  CNT    = (float*)(ws + off);  off += (size_t)NIMG * 4;
  bf16_t* MEANBF = (bf16_t*)(ws + off); off += SZ_IMG_BF;
  float*  OUTA   = (float*)(ws + off);  off += SZ_IMG_F32;
  float*  ACC2   = (float*)(ws + off);  off += SZ_CLS_F32;
  float*  OUTC   = (float*)(ws + off);  off += SZ_CLS_F32;
  bf16_t* WREN   = (bf16_t*)(ws + off); off += (size_t)12 * C_DIM * C_DIM * 2;
  bf16_t* WOUT   = (bf16_t*)(ws + off); off += (size_t)2 * C_DIM * C_DIM * 2;
  float*  OUTB   = ACC;

  const dim3 tb256(256);
  const dim3 tgrid(IMG_W / 32, C_DIM / 32, IMG_B * IMG_H);
  const dim3 tblk(32, 8);

  // weight prep
  k_renorm<<<192, tb256, 0, stream>>>(Wl, Wr, WREN);
  k_wout_cvt<<<(2 * 65536) / 256, tb256, 0, stream>>>(Wout_img, Wout_cls, WOUT);

  // activation staging
  k_nchw_to_nodes<<<tgrid, tblk, 0, stream>>>(x, XI_F32, XI_BF);
  k_cls_init<<<(NCLS * C_DIM) / 256, tb256, 0, stream>>>(class_x, XC_F32, XC_BF);

  for (int l = 0; l < 2; ++l) {
    const bf16_t* wl0 = WREN + (size_t)((l * 3 + 0) * 2 + 0) * C_DIM * C_DIM;
    const bf16_t* wr0 = WREN + (size_t)((l * 3 + 0) * 2 + 1) * C_DIM * C_DIM;
    const bf16_t* wl1 = WREN + (size_t)((l * 3 + 1) * 2 + 0) * C_DIM * C_DIM;
    const bf16_t* wr1 = WREN + (size_t)((l * 3 + 1) * 2 + 1) * C_DIM * C_DIM;
    const bf16_t* wl2 = WREN + (size_t)((l * 3 + 2) * 2 + 0) * C_DIM * C_DIM;
    const bf16_t* wr2 = WREN + (size_t)((l * 3 + 2) * 2 + 1) * C_DIM * C_DIM;
    const float* bl0 = bl + (size_t)(l * 3 + 0) * C_DIM;
    const float* bl1 = bl + (size_t)(l * 3 + 1) * C_DIM;
    const float* bl2 = bl + (size_t)(l * 3 + 2) * C_DIM;

    // --- image<-image: mean over ei_ii, GEMM A -> OUTA ---
    hipMemsetAsync(ACC, 0, SZ_IMG_F32, stream);
    hipMemsetAsync(CNT, 0, (size_t)NIMG * 4, stream);
    k_scatter<<<(E_II * 32 + 255) / 256, tb256, 0, stream>>>(
        XI_F32, ei_ii, ei_ii + E_II, E_II, ACC, CNT);
    k_mean_bf16<<<NIMG, tb256, 0, stream>>>(ACC, CNT, MEANBF);
    k_gemm_sage<<<dim3(NIMG / 32), tb256, 0, stream>>>(
        MEANBF, XI_BF, wl0, wr0, bl0, OUTA);

    // --- image<-class: mean over ei_ci, GEMM B -> OUTB (reuses ACC region) ---
    hipMemsetAsync(ACC, 0, SZ_IMG_F32, stream);
    hipMemsetAsync(CNT, 0, (size_t)NIMG * 4, stream);
    k_scatter<<<(E_CI * 32 + 255) / 256, tb256, 0, stream>>>(
        XC_F32, ei_ci, ei_ci + E_CI, E_CI, ACC, CNT);
    k_mean_bf16<<<NIMG, tb256, 0, stream>>>(ACC, CNT, MEANBF);
    k_gemm_sage<<<dim3(NIMG / 32), tb256, 0, stream>>>(
        MEANBF, XI_BF, wl1, wr1, bl1, OUTB);

    // --- class<-image: mean over ei_ic, GEMM C -> OUTC ---
    hipMemsetAsync(ACC2, 0, SZ_CLS_F32, stream);
    hipMemsetAsync(CNT, 0, (size_t)NCLS * 4, stream);
    k_scatter<<<(E_IC * 32 + 255) / 256, tb256, 0, stream>>>(
        XI_F32, ei_ic, ei_ic + E_IC, E_IC, ACC2, CNT);
    k_mean_bf16<<<NCLS, tb256, 0, stream>>>(ACC2, CNT, MEANBF);
    k_gemm_sage<<<dim3(NCLS / 32), tb256, 0, stream>>>(
        MEANBF, XC_BF, wl2, wr2, bl2, OUTC);

    // --- normalize + sum + mp_silu -> next-layer activations ---
    k_combine2<<<NIMG, tb256, 0, stream>>>(OUTA, OUTB, XI_F32, XI_BF);
    k_combine1<<<NCLS, tb256, 0, stream>>>(OUTC, XC_F32, XC_BF);
  }

  // final per-type linear
  k_gemm_sage<<<dim3(NIMG / 32), tb256, 0, stream>>>(
      XI_BF, nullptr, WOUT, nullptr, bout_img, OUTA);
  k_nodes_to_nchw<<<tgrid, tblk, 0, stream>>>(OUTA, (float*)d_out);
  k_gemm_sage<<<dim3(NCLS / 32), tb256, 0, stream>>>(
      XC_BF, nullptr, WOUT + (size_t)C_DIM * C_DIM, nullptr, bout_cls,
      (float*)d_out + (size_t)IMG_B * C_DIM * IMG_H * IMG_W);
}
